// DLinear_436
// MI455X (gfx1250) — compile-verified, matrix operand order checked by reading, not run on hardware
//
#include <hip/hip_runtime.h>

typedef __attribute__((ext_vector_type(16))) _Float16 v16h;
typedef __attribute__((ext_vector_type(8)))  _Float16 v8h;
typedef __attribute__((ext_vector_type(8)))  float    v8f;

#define B_    32
#define T_IN_ 12
#define H_    64
#define W_    64
#define C_IN_ 4
#define O_    48
#define K_    8
#define S_    4
#define NH_   15
#define NW_   15
#define L_    225      // NH*NW
#define KDIM_ 96       // 2 * T_IN * C_IN  (res block then trend block)
#define MROWS_ 7200    // B * L
#define MTILES_ 450    // MROWS/16
#define COMB_BYTES (32u*64u*64u*96u*2u)   // 25,165,824 (256B aligned)

// ---------------- zero the atomic-add target ----------------
__global__ void zero_out(float* __restrict__ out, int n) {
    int i = blockIdx.x * 256 + threadIdx.x;
    if (i < n) out[i] = 0.0f;
}

// ---------------- closed-form decomposition -> f16 staging ----------------
// COMB[pix][k]: k<48 -> res(t,c), k>=48 -> trend(t,c), k(t,c)=t*4+c, pix=b*4096+h*64+w
__global__ void prep_comb(const float* __restrict__ x, _Float16* __restrict__ comb) {
    int pix = blockIdx.x * 256 + threadIdx.x;          // 0 .. 131071
    int b  = pix >> 12;
    int hw = pix & 4095;
    const float* xb = x + (size_t)b * T_IN_ * 4096 * 4 + (size_t)hw * 4;
    float xs[T_IN_][C_IN_];
    float s[C_IN_] = {0.f, 0.f, 0.f, 0.f};
    #pragma unroll
    for (int t = 0; t < T_IN_; ++t) {
        float4 v = *(const float4*)(xb + (size_t)t * 4096 * 4);
        xs[t][0] = v.x; xs[t][1] = v.y; xs[t][2] = v.z; xs[t][3] = v.w;
        s[0] += v.x; s[1] += v.y; s[2] += v.z; s[3] += v.w;
    }
    _Float16* cp = comb + (size_t)pix * KDIM_;
    #pragma unroll
    for (int t = 0; t < T_IN_; ++t) {
        #pragma unroll
        for (int c = 0; c < C_IN_; ++c) {
            // MA window always covers all 12 real frames + (12-t) copies of x0 + (t+1) copies of x11
            float tr = (s[c] + (float)(T_IN_ - t) * xs[0][c] + (float)(t + 1) * xs[T_IN_ - 1][c]) * (1.0f / 25.0f);
            cp[t * 4 + c]      = (_Float16)(xs[t][c] - tr);   // res  -> pairs with w_s
            cp[48 + t * 4 + c] = (_Float16)tr;                // trend-> pairs with w_t
        }
    }
}

// ---------------- pre-swizzle weights into per-lane B-matrix layout ----------------
// wpack[((ij*9 + nt*3 + kc)*32 + lane)*16 + e]
// B-tile lane mapping (16x16x32 f16): N = lane&15; element e -> K = ((e&8)<<1) + ((lane>>4)<<3) + (e&7)
__global__ void prep_w(const float* __restrict__ w_s, const float* __restrict__ w_t,
                       _Float16* __restrict__ wpack) {
    int idx = blockIdx.x * 256 + threadIdx.x;          // 0 .. 294911
    int e    = idx & 15;
    int lane = (idx >> 4) & 31;
    int r    = idx >> 9;                               // ij*9 + nt*3 + kc
    int kc   = r % 3;  r /= 3;
    int nt   = r % 3;  r /= 3;
    int ij   = r;                                      // i*8 + j
    int k = kc * 32 + ((e & 8) << 1) + ((lane >> 4) << 3) + (e & 7);   // 0..95
    int o = nt * 16 + (lane & 15);                                     // 0..47
    // w[o,t,c,i,j] flat = (o*48 + (t*4+c))*64 + ij
    float v = (k < 48) ? w_s[(o * 48 + k) * 64 + ij]
                       : w_t[(o * 48 + (k - 48)) * 64 + ij];
    wpack[idx] = (_Float16)v;
}

// ---------------- WMMA GEMM + fused fold (overlap-add) ----------------
// one wave = one (16-row M-tile, ij) pair; 3 k-chunks x 3 n-tiles = 9 v_wmma_f32_16x16x32_f16
__global__ void __launch_bounds__(256) gemm_fold(const _Float16* __restrict__ comb,
                                                 const _Float16* __restrict__ wpack,
                                                 const float* __restrict__ b_s,
                                                 const float* __restrict__ b_t,
                                                 float* __restrict__ out) {
    const int lane   = threadIdx.x & 31;
    const int gwave  = blockIdx.x * 8 + (threadIdx.x >> 5);  // 0 .. 28799
    const int ij     = gwave / MTILES_;                      // 0..63
    const int mtile  = gwave % MTILES_;                      // 0..449
    const int i      = ij >> 3, j = ij & 7;
    const int lanelo = lane & 15, lanehi = lane >> 4;

    // A-matrix: lane holds row M = lane&15; K-halves split by lane>>4 (ISA 16-bit A 16x32 layout)
    int row = mtile * 16 + lanelo;                 // 0..7199
    int b   = row / L_;  int l  = row - b * L_;
    int lh  = l / NW_;   int lw = l - lh * NW_;
    int h   = lh * S_ + i, w = lw * S_ + j;
    const _Float16* ap = comb + ((size_t)((b * H_ + h) * W_ + w)) * KDIM_;

    v8f acc[3];
    acc[0] = v8f{}; acc[1] = v8f{}; acc[2] = v8f{};

    #pragma unroll
    for (int kc = 0; kc < 3; ++kc) {
        union { v16h v; v8h h8[2]; } A;
        int kbase = kc * 32 + lanehi * 8;
        A.h8[0] = *(const v8h*)(ap + kbase);        // K = kbase .. kbase+7
        A.h8[1] = *(const v8h*)(ap + kbase + 16);   // K = kbase+16 .. kbase+23
        const _Float16* wp = wpack + ((size_t)ij * 9 + kc) * 32 * 16 + (size_t)lane * 16;
        v16h b0 = *(const v16h*)(wp);               // nt = 0
        v16h b1 = *(const v16h*)(wp + 3 * 32 * 16); // nt = 1
        v16h b2 = *(const v16h*)(wp + 6 * 32 * 16); // nt = 2
        acc[0] = __builtin_amdgcn_wmma_f32_16x16x32_f16(false, A.v, false, b0, (short)0, acc[0], false, false);
        acc[1] = __builtin_amdgcn_wmma_f32_16x16x32_f16(false, A.v, false, b1, (short)0, acc[1], false, false);
        acc[2] = __builtin_amdgcn_wmma_f32_16x16x32_f16(false, A.v, false, b2, (short)0, acc[2], false, false);
    }

    // D layout: lane holds N = lane&15; VGPR r holds M = r + 8*(lane>>4).
    // This lane's 8 rows are CONSECUTIVE: mtile*16 + lanehi*8 + r. Decode the first
    // once, then strength-reduce with branchless carry updates (lw->lh->b wraps).
    int row0 = mtile * 16 + lanehi * 8;
    int b2  = row0 / L_;  int l2  = row0 - b2 * L_;
    int lh2 = l2 / NW_;   int lw2 = l2 - lh2 * NW_;
    int pixpart[8];
    #pragma unroll
    for (int r = 0; r < 8; ++r) {
        int h2 = lh2 * S_ + i, w2 = lw2 * S_ + j;
        // out[b][to][h][w][co]: addr = b*12*64*64*4 + to*64*64*4 + h*64*4 + w*4 + co
        pixpart[r] = b2 * (12 * H_ * W_ * 4) + h2 * (W_ * 4) + w2 * 4;
        ++lw2;
        if (lw2 == NW_) {               // carry into lh
            lw2 = 0; ++lh2;
            if (lh2 == NH_) { lh2 = 0; ++b2; }  // carry into b
        }
    }

    #pragma unroll
    for (int nt = 0; nt < 3; ++nt) {
        int o = nt * 16 + lanelo;
        float bias = b_s[o * 64 + ij] + b_t[o * 64 + ij];
        int ooff = (o >> 2) * (H_ * W_ * 4) + (o & 3);   // to*16384 + co
        #pragma unroll
        for (int r = 0; r < 8; ++r) {
            atomicAdd(out + pixpart[r] + ooff, acc[nt][r] + bias);
        }
    }
}

extern "C" void kernel_launch(void* const* d_in, const int* in_sizes, int n_in,
                              void* d_out, int out_size, void* d_ws, size_t ws_size,
                              hipStream_t stream) {
    (void)in_sizes; (void)n_in; (void)ws_size;
    const float* x   = (const float*)d_in[0];
    const float* w_s = (const float*)d_in[1];
    const float* b_s = (const float*)d_in[2];
    const float* w_t = (const float*)d_in[3];
    const float* b_t = (const float*)d_in[4];
    float* out = (float*)d_out;

    _Float16* comb  = (_Float16*)d_ws;                              // 25,165,824 B
    _Float16* wpack = (_Float16*)((char*)d_ws + COMB_BYTES);        //    589,824 B

    zero_out <<<(out_size + 255) / 256, 256, 0, stream>>>(out, out_size);
    prep_comb<<<512,  256, 0, stream>>>(x, comb);                   // 131072 pixels
    prep_w   <<<1152, 256, 0, stream>>>(w_s, w_t, wpack);           // 294912 elements
    gemm_fold<<<3600, 256, 0, stream>>>(comb, wpack, b_s, b_t, out);// 28800 waves
}
